// CadenceGNNPytorch_72335839199439
// MI455X (gfx1250) — compile-verified, matrix operand order checked by reading, not run on hardware
//
#include <hip/hip_runtime.h>
#include <hip/hip_bf16.h>
#include <math.h>

// ---------------------------------------------------------------------------
// CadenceGNN forward for MI455X (gfx1250, wave32, WMMA + TDM).
// Dense layers: bf16 WMMA (v_wmma_f32_16x16x32_bf16) with f32 accumulation.
// At 23.3 TB/s HBM the GEMMs are memory-bound only if matrix throughput is
// >=1.5 PFLOPS, which rules out the f32 16x16x4 path (8x lower K-depth) and
// makes bf16 the right precision. Weight tiles are staged into LDS by the
// Tensor Data Mover (tensor_load_to_lds, one 2D-tile descriptor per block,
// waited with s_wait_tensorcnt) so all 8 waves stay free during the fill.
// bf16 packing uses one v_perm_b32 per pair (round-to-nearest bias add).
// Edge aggregation: float4 gather + global_atomic_add_f32 scatter.
// Workspace requirement: ~490 MB.
// ---------------------------------------------------------------------------

#define GN_N     100000
#define GN_E     400000
#define GN_R     4
#define GN_DIN   128
#define GN_H     256
#define GN_H2    128
#define GN_H4    64
#define GN_OUT   3
#define GN_CMAX  25000
#define GN_EPS   1e-5f

typedef __attribute__((ext_vector_type(16))) __bf16 v16bf;
typedef __attribute__((ext_vector_type(8)))  float  v8f;
typedef __attribute__((ext_vector_type(4)))  unsigned int u32x4;
typedef __attribute__((ext_vector_type(8)))  int          i32x8;
typedef __attribute__((ext_vector_type(4)))  int          i32x4;

union AFrag { unsigned int u[8]; v16bf v; };

// Round-to-nearest bf16 pair pack: 2x bias add + 1x v_perm_b32.
__device__ __forceinline__ unsigned int pack2_bf16(float a, float b) {
    unsigned int x = __builtin_bit_cast(unsigned int, a) + 0x8000u;
    unsigned int y = __builtin_bit_cast(unsigned int, b) + 0x8000u;
    return __builtin_amdgcn_perm(y, x, 0x07060302u);  // {y.b3,y.b2,x.b3,x.b2}
}

// ---- pack fp32 weight [Krows x Ncols] into bf16 K-pair dwords -------------
__global__ void pack_w_kernel(const float* __restrict__ W,
                              unsigned int* __restrict__ P,
                              int Krows, int Ncols) {
    int idx = blockIdx.x * blockDim.x + threadIdx.x;
    int total = (Krows >> 1) * Ncols;
    if (idx >= total) return;
    int kp = idx / Ncols, n = idx - kp * Ncols;
    P[idx] = pack2_bf16(W[(size_t)(2 * kp) * Ncols + n],
                        W[(size_t)(2 * kp + 1) * Ncols + n]);
}

// ---- bf16 WMMA GEMM: Y[M,Nout] = act(X[M,Kin] @ W + bias [+ Y]) -----------
// Kin = KC*32 (compile time), per-block N tile = NTILES*16.
// Block = 256 threads = 8 waves; each wave owns a 16-row strip.
template <int KC, int NTILES>
__global__ __launch_bounds__(256)
void gemm_bf16_kernel(const float* __restrict__ X,
                      const unsigned int* __restrict__ Wp,  // (Kin/2) x Nout
                      const float* __restrict__ bias,
                      float* __restrict__ Y,
                      int M, int Nout, int act, int accum) {
    constexpr int Kin = KC * 32;
    constexpr int NT  = NTILES * 16;
    __shared__ unsigned int ldsW[(Kin / 2) * NT];

    const int tid    = threadIdx.x;
    const int n_base = blockIdx.y * NT;

#if defined(__gfx1250__) && __has_builtin(__builtin_amdgcn_tensor_load_to_lds)
    // TDM: one 2D-tile descriptor moves the (Kin/2) x NT dword tile
    // (row stride Nout dwords) from global into LDS. Wave 0 issues and
    // waits TENSORcnt; the workgroup barrier publishes it to all waves.
    if (tid < 32) {
        unsigned int lds_off = (unsigned int)(uintptr_t)(void*)ldsW; // addr[31:0] = LDS offset
        unsigned long long ga = (unsigned long long)(uintptr_t)(Wp + n_base);
        u32x4 g0;
        g0.x = 1u;                                   // count=1, user mode, no gather
        g0.y = lds_off;                              // lds_addr [63:32]
        g0.z = (unsigned int)(ga & 0xFFFFFFFFu);     // global_addr [95:64]
        g0.w = (unsigned int)((ga >> 32) & 0x01FFFFFFu) | (2u << 30); // [120:96] + type=2
        i32x8 g1;
        g1[0] = (int)(2u << 16);                     // workgroup_mask=0, data_size=4B
        g1[1] = (int)(((unsigned int)Nout & 0xFFFFu) << 16);          // tensor_dim0 lo16
        g1[2] = (int)((((unsigned int)Nout >> 16) & 0xFFFFu)          // tensor_dim0 hi16
                      | (((unsigned int)(Kin / 2) & 0xFFFFu) << 16)); // tensor_dim1 lo16
        g1[3] = (int)((((unsigned int)(Kin / 2) >> 16) & 0xFFFFu)     // tensor_dim1 hi16
                      | ((unsigned int)NT << 16));                    // tile_dim0
        g1[4] = (int)(unsigned int)(Kin / 2);        // tile_dim1 (tile_dim2=0)
        g1[5] = (int)(unsigned int)Nout;             // tensor_dim0_stride lo32
        g1[6] = 0;                                   // stride hi16, dim1_stride lo16
        g1[7] = 0;
        i32x4 gz4 = {0, 0, 0, 0};                    // groups 2/3 unused (2D tile)
        i32x8 gz8 = {0, 0, 0, 0, 0, 0, 0, 0};
        __builtin_amdgcn_tensor_load_to_lds(g0, g1, gz4, gz4, gz8, 0);
        __builtin_amdgcn_s_wait_tensorcnt((short)0);
    }
#else
    for (int idx = tid; idx < (Kin / 2) * NT; idx += 256) {
        int kp = idx / NT, nl = idx - kp * NT;
        ldsW[idx] = Wp[(size_t)kp * Nout + n_base + nl];
    }
#endif
    __syncthreads();

    const int wave = tid >> 5;
    const int lane = tid & 31;
    const int half = lane >> 4;   // lanes 16-31 hold the +8 K-slice / +8 rows
    const int l16  = lane & 15;
    const int row0 = blockIdx.x * 128 + wave * 16;
    if (row0 >= M) return;        // wave-uniform; EXEC stays all-ones for WMMA

    const int m  = row0 + l16;
    const int mc = (m < M) ? m : (M - 1);   // clamp loads for partial tiles
    const float* xrow = X + (size_t)mc * Kin;
    __builtin_prefetch(xrow, 0, 1);         // global_prefetch_b8

    v8f zero = {};
    v8f acc[NTILES];
#pragma unroll
    for (int t = 0; t < NTILES; ++t) acc[t] = zero;

#pragma unroll
    for (int c = 0; c < KC; ++c) {
        // A fragment, 16x32 bf16 per ISA layout:
        // lane<16 -> K in {0..7,16..23}; lane>=16 -> +8.
        AFrag a;
#pragma unroll
        for (int i = 0; i < 8; ++i) {
            int k = (c << 5) + ((i & 3) << 1) + (half << 3) + ((i >> 2) << 4);
            a.u[i] = pack2_bf16(xrow[k], xrow[k + 1]);
        }
#pragma unroll
        for (int t = 0; t < NTILES; ++t) {
            AFrag b;   // B fragment 32x16, mirrored layout, from LDS pairs
#pragma unroll
            for (int i = 0; i < 8; ++i) {
                int k = (c << 5) + ((i & 3) << 1) + (half << 3) + ((i >> 2) << 4);
                b.u[i] = ldsW[(k >> 1) * NT + t * 16 + l16];
            }
            acc[t] = __builtin_amdgcn_wmma_f32_16x16x32_bf16(
                false, a.v, false, b.v, (short)0, acc[t], false, false);
        }
    }

    // Epilogue: C/D layout -> VGPR j holds row row0 + j + 8*half, col = l16.
#pragma unroll
    for (int t = 0; t < NTILES; ++t) {
        int n = n_base + t * 16 + l16;
        float bv = bias ? bias[n] : 0.0f;
#pragma unroll
        for (int j = 0; j < 8; ++j) {
            int mm = row0 + j + (half << 3);
            if (mm < M) {
                size_t o = (size_t)mm * Nout + n;
                float v = acc[t][j] + bv;
                if (accum) v += Y[o];
                if (act) v = fmaxf(v, 0.0f);
                Y[o] = v;
            }
        }
    }
}

// ---- scatter-mean numerator: acc[dst] += feat[src], cnt[dst] += 1 ---------
__global__ void scatter_add_kernel(const float* __restrict__ feat,
                                   const int* __restrict__ src,  // null => identity
                                   const int* __restrict__ dst,
                                   float* __restrict__ acc,
                                   float* __restrict__ cnt,
                                   int nE, int F) {
    int f4  = F >> 2;
    int idx = blockIdx.x * blockDim.x + threadIdx.x;
    if (idx >= nE * f4) return;
    int e = idx / f4;
    int c = (idx - e * f4) << 2;
    int s = src ? src[e] : e;
    int d = dst[e];
    const float4 v = *reinterpret_cast<const float4*>(feat + (size_t)s * F + c);
    float* o = acc + (size_t)d * F + c;
    atomicAdd(o + 0, v.x); atomicAdd(o + 1, v.y);
    atomicAdd(o + 2, v.z); atomicAdd(o + 3, v.w);
    if (c == 0) atomicAdd(cnt + d, 1.0f);
}

__global__ void seg_div_kernel(float* __restrict__ acc,
                               const float* __restrict__ cnt,
                               int total, int F) {
    int idx = blockIdx.x * blockDim.x + threadIdx.x;
    if (idx >= total) return;
    acc[idx] /= fmaxf(cnt[idx / F], 1.0f);
}

// ---- per-row L2-normalize then accumulate across relations ----------------
__global__ void l2norm_acc_kernel(const float* __restrict__ o,
                                  float* __restrict__ acc,
                                  int F, int init) {
    __shared__ float red[256];
    const int row = blockIdx.x, tid = threadIdx.x;
    float v = (tid < F) ? o[(size_t)row * F + tid] : 0.0f;
    red[tid] = v * v;
    __syncthreads();
    for (int s = 128; s > 0; s >>= 1) {
        if (tid < s) red[tid] += red[tid + s];
        __syncthreads();
    }
    float inv = 1.0f / fmaxf(sqrtf(red[0]), 1e-12f);
    if (tid < F) {
        size_t idx = (size_t)row * F + tid;
        float nv = v * inv;
        acc[idx] = init ? nv : acc[idx] + nv;
    }
}

__global__ void acc_kernel(const float* __restrict__ o, float* __restrict__ acc,
                           int total, int init) {
    int idx = blockIdx.x * blockDim.x + threadIdx.x;
    if (idx >= total) return;
    acc[idx] = init ? o[idx] : acc[idx] + o[idx];
}

// ---- out = LayerNorm(relu?(in*scale)) * g + b  (in==out allowed) ----------
__global__ void row_ln_kernel(const float* __restrict__ in, float* __restrict__ out,
                              const float* __restrict__ g, const float* __restrict__ b,
                              int F, float scale, int relu) {
    __shared__ float s1[256], s2[256];
    const int row = blockIdx.x, tid = threadIdx.x;
    float v = 0.0f;
    if (tid < F) {
        v = in[(size_t)row * F + tid] * scale;
        if (relu) v = fmaxf(v, 0.0f);
    }
    s1[tid] = (tid < F) ? v : 0.0f;
    s2[tid] = (tid < F) ? v * v : 0.0f;
    __syncthreads();
    for (int s = 128; s > 0; s >>= 1) {
        if (tid < s) { s1[tid] += s1[tid + s]; s2[tid] += s2[tid + s]; }
        __syncthreads();
    }
    float mu   = s1[0] / F;
    float var  = s2[0] / F - mu * mu;
    float rstd = rsqrtf(fmaxf(var, 0.0f) + GN_EPS);
    if (tid < F) out[(size_t)row * F + tid] = (v - mu) * rstd * g[tid] + b[tid];
}

// ---- eval-mode BatchNorm: h = h * g/sqrt(1+eps) + b, in place -------------
__global__ void bn_kernel(float* __restrict__ h, const float* __restrict__ g,
                          const float* __restrict__ b, int total, int F) {
    int idx = blockIdx.x * blockDim.x + threadIdx.x;
    if (idx >= total) return;
    int f = idx % F;
    h[idx] = h[idx] * (g[f] * rsqrtf(1.0f + GN_EPS)) + b[f];
}

// ---- out[i,:] = table[seg[i],:] -------------------------------------------
__global__ void gather_rows_kernel(const float* __restrict__ table,
                                   const int* __restrict__ seg,
                                   float* __restrict__ out, int total, int F) {
    int idx = blockIdx.x * blockDim.x + threadIdx.x;
    if (idx >= total) return;
    int i = idx / F, f = idx - i * F;
    out[idx] = table[(size_t)seg[i] * F + f];
}

// ---- fused classifier tail: logits = t@W2[64x3]+b2, softmax ---------------
__global__ void clf_softmax_kernel(const float* __restrict__ t,
                                   const float* __restrict__ W2,
                                   const float* __restrict__ b2,
                                   float* __restrict__ out, int M) {
    int i = blockIdx.x * blockDim.x + threadIdx.x;
    if (i >= M) return;
    const float* ti = t + (size_t)i * GN_H4;
    float l0 = b2[0], l1 = b2[1], l2 = b2[2];
#pragma unroll 8
    for (int k = 0; k < GN_H4; ++k) {
        float v = ti[k];
        l0 += v * W2[k * 3 + 0];
        l1 += v * W2[k * 3 + 1];
        l2 += v * W2[k * 3 + 2];
    }
    float mx = fmaxf(l0, fmaxf(l1, l2));
    float e0 = __expf(l0 - mx), e1 = __expf(l1 - mx), e2 = __expf(l2 - mx);
    float s = e0 + e1 + e2;
    out[(size_t)i * 3 + 0] = e0 / s;
    out[(size_t)i * 3 + 1] = e1 / s;
    out[(size_t)i * 3 + 2] = e2 / s;
}

// ---------------------------------------------------------------------------
static void gemm_launch(const float* X, const unsigned int* Wp, const float* bias,
                        float* Y, int M, int Kin, int Nout, int act, int accum,
                        hipStream_t s) {
    dim3 blk(256);
    if (Kin == 128 && Nout >= 128) {
        dim3 g((M + 127) / 128, Nout / 128);
        gemm_bf16_kernel<4, 8><<<g, blk, 0, s>>>(X, Wp, bias, Y, M, Nout, act, accum);
    } else if (Kin == 256) {
        dim3 g((M + 127) / 128, Nout / 128);
        gemm_bf16_kernel<8, 8><<<g, blk, 0, s>>>(X, Wp, bias, Y, M, Nout, act, accum);
    } else {  // Kin == 128, Nout == 64
        dim3 g((M + 127) / 128, 1);
        gemm_bf16_kernel<4, 4><<<g, blk, 0, s>>>(X, Wp, bias, Y, M, Nout, act, accum);
    }
}

extern "C" void kernel_launch(void* const* d_in, const int* in_sizes, int n_in,
                              void* d_out, int out_size, void* d_ws, size_t ws_size,
                              hipStream_t stream) {
    const int N = GN_N, E = GN_E, R = GN_R, DIN = GN_DIN, H = GN_H;
    const int H2 = GN_H2, H4 = GN_H4, CMAX = GN_CMAX;

    // Inputs in setup_inputs() order: x, then params leaves, then edge_index, onset_div.
    auto f = [&](int i) { return (const float*)d_in[i]; };
    const float* x        = f(0);
    const float* proj_W   = f(1);  const float* proj_b   = f(2);
    const float* l0_Wl    = f(3);  const float* l0_bl    = f(4);  const float* l0_Wr = f(5);
    const float* l1_Wl    = f(6);  const float* l1_bl    = f(7);  const float* l1_Wr = f(8);
    const float* ln0_g    = f(9);  const float* ln0_b    = f(10);
    const float* ln1_g    = f(11); const float* ln1_b    = f(12);
    const float* mlp_W1   = f(13); const float* mlp_b1   = f(14);
    const float* mlp_bn_g = f(15); const float* mlp_bn_b = f(16);
    const float* mlp_W2   = f(17); const float* mlp_b2   = f(18);
    const float* norm_g   = f(19); const float* norm_b   = f(20);
    const float* pool_W1  = f(21); const float* pool_b1  = f(22);
    const float* pool_ln_g= f(23); const float* pool_ln_b= f(24);
    const float* pool_W2  = f(25); const float* pool_b2  = f(26);
    const float* clf_W1   = f(27); const float* clf_b1   = f(28);
    const float* clf_bn_g = f(29); const float* clf_bn_b = f(30);
    const float* clf_W2   = f(31); const float* clf_b2   = f(32);
    const int*   eidx     = (const int*)d_in[33];
    const int*   onset    = (const int*)d_in[34];

    // -------- workspace carve (~490 MB fp32) --------
    float* wsf = (float*)d_ws;
    size_t off = 0;
    auto carve = [&](size_t n) { float* p = wsf + off; off += (n + 3) & ~(size_t)3; return p; };
    float* bufM   = carve((size_t)N * H);      // messages / h1 / broadcast
    float* bufO   = carve((size_t)N * H);      // per-relation output / mlp tmp
    float* bufAcc = carve((size_t)N * H);      // relation-mean accumulator
    float* bufH   = carve((size_t)N * H);      // h after layer 0
    float* bufXP  = carve((size_t)N * DIN);    // projected feats / mlp out
    float* cnt    = carve(N);
    float* pooled = carve((size_t)CMAX * H2);
    float* pooledT= carve((size_t)CMAX * H2);
    float* pcnt   = carve(CMAX);

    unsigned int* wp = (unsigned int*)(wsf + off);
    unsigned int* projWp = wp;
    unsigned int* l0Wlp  = projWp + (size_t)R * (DIN / 2) * DIN;
    unsigned int* l0Wrp  = l0Wlp  + (size_t)R * (DIN / 2) * H;
    unsigned int* l1Wlp  = l0Wrp  + (size_t)R * (DIN / 2) * H;
    unsigned int* l1Wrp  = l1Wlp  + (size_t)R * (H / 2) * H;
    unsigned int* mlpW1p = l1Wrp  + (size_t)R * (H / 2) * H;
    unsigned int* mlpW2p = mlpW1p + (size_t)(H / 2) * H;
    unsigned int* poolW1p= mlpW2p + (size_t)(H / 2) * H2;
    unsigned int* poolW2p= poolW1p+ (size_t)(H2 / 2) * H2;
    unsigned int* clfW1p = poolW2p+ (size_t)(H2 / 2) * H2;

    auto pack = [&](const float* W, unsigned int* P, int K, int Nn) {
        int total = (K / 2) * Nn;
        pack_w_kernel<<<(total + 255) / 256, 256, 0, stream>>>(W, P, K, Nn);
    };
    pack(proj_W, projWp, R * DIN, DIN);
    pack(l0_Wl,  l0Wlp,  R * DIN, H);
    pack(l0_Wr,  l0Wrp,  R * DIN, H);
    pack(l1_Wl,  l1Wlp,  R * H,   H);
    pack(l1_Wr,  l1Wrp,  R * H,   H);
    pack(mlp_W1, mlpW1p, H,  H);
    pack(mlp_W2, mlpW2p, H,  H2);
    pack(pool_W1, poolW1p, H2, H2);
    pack(pool_W2, poolW2p, H2, H2);
    pack(clf_W1,  clfW1p,  H2, H4);

    // -------- layer 0: per-relation SAGEConv(project, normalize) --------
    for (int r = 0; r < R; ++r) {
        gemm_launch(x, projWp + (size_t)r * (DIN / 2) * DIN, proj_b + r * DIN,
                    bufXP, N, DIN, DIN, /*relu*/1, 0, stream);
        (void)hipMemsetAsync(bufM, 0, (size_t)N * DIN * sizeof(float), stream);
        (void)hipMemsetAsync(cnt,  0, (size_t)N * sizeof(float), stream);
        const int* src = eidx + (size_t)r * 2 * E;
        const int* dst = src + E;
        int tot = E * (DIN / 4);
        scatter_add_kernel<<<(tot + 255) / 256, 256, 0, stream>>>(bufXP, src, dst, bufM, cnt, E, DIN);
        int totM = N * DIN;
        seg_div_kernel<<<(totM + 255) / 256, 256, 0, stream>>>(bufM, cnt, totM, DIN);
        gemm_launch(bufM, l0Wlp + (size_t)r * (DIN / 2) * H, l0_bl + r * H,
                    bufO, N, DIN, H, 0, 0, stream);
        gemm_launch(x, l0Wrp + (size_t)r * (DIN / 2) * H, nullptr,
                    bufO, N, DIN, H, 0, /*accum*/1, stream);
        l2norm_acc_kernel<<<N, 256, 0, stream>>>(bufO, bufAcc, H, r == 0);
    }
    row_ln_kernel<<<N, 256, 0, stream>>>(bufAcc, bufH, ln0_g, ln0_b, H, 1.0f / R, 1);

    // -------- layer 1: per-relation plain SAGEConv --------
    for (int r = 0; r < R; ++r) {
        (void)hipMemsetAsync(bufM, 0, (size_t)N * H * sizeof(float), stream);
        (void)hipMemsetAsync(cnt,  0, (size_t)N * sizeof(float), stream);
        const int* src = eidx + (size_t)r * 2 * E;
        const int* dst = src + E;
        int tot = E * (H / 4);
        scatter_add_kernel<<<(tot + 255) / 256, 256, 0, stream>>>(bufH, src, dst, bufM, cnt, E, H);
        int totM = N * H;
        seg_div_kernel<<<(totM + 255) / 256, 256, 0, stream>>>(bufM, cnt, totM, H);
        gemm_launch(bufM, l1Wlp + (size_t)r * (H / 2) * H, l1_bl + r * H,
                    bufO, N, H, H, 0, 0, stream);
        gemm_launch(bufH, l1Wrp + (size_t)r * (H / 2) * H, nullptr,
                    bufO, N, H, H, 0, 1, stream);
        acc_kernel<<<(N * H + 255) / 256, 256, 0, stream>>>(bufO, bufAcc, N * H, r == 0);
    }
    row_ln_kernel<<<N, 256, 0, stream>>>(bufAcc, bufM, ln1_g, ln1_b, H, 1.0f / R, 1);

    // -------- GNN.mlp --------
    gemm_launch(bufM, mlpW1p, mlp_b1, bufO, N, H, H, 1, 0, stream);
    bn_kernel<<<(N * H + 255) / 256, 256, 0, stream>>>(bufO, mlp_bn_g, mlp_bn_b, N * H, H);
    gemm_launch(bufO, mlpW2p, mlp_b2, bufXP, N, H, H2, 0, 0, stream);

    // -------- onset pooling + pool MLP --------
    (void)hipMemsetAsync(pooled, 0, (size_t)CMAX * H2 * sizeof(float), stream);
    (void)hipMemsetAsync(pcnt,   0, (size_t)CMAX * sizeof(float), stream);
    int totP = N * (H2 / 4);
    scatter_add_kernel<<<(totP + 255) / 256, 256, 0, stream>>>(bufXP, nullptr, onset, pooled, pcnt, N, H2);
    int totC = CMAX * H2;
    seg_div_kernel<<<(totC + 255) / 256, 256, 0, stream>>>(pooled, pcnt, totC, H2);
    row_ln_kernel<<<CMAX, 256, 0, stream>>>(pooled, pooled, norm_g, norm_b, H2, 1.0f, 0);
    gemm_launch(pooled, poolW1p, pool_b1, pooledT, CMAX, H2, H2, 1, 0, stream);
    row_ln_kernel<<<CMAX, 256, 0, stream>>>(pooledT, pooledT, pool_ln_g, pool_ln_b, H2, 1.0f, 0);
    gemm_launch(pooledT, poolW2p, pool_b2, pooled, CMAX, H2, H2, 0, 0, stream);
    gather_rows_kernel<<<(N * H2 + 255) / 256, 256, 0, stream>>>(pooled, onset, bufM, N * H2, H2);

    // -------- classifier + softmax --------
    gemm_launch(bufM, clfW1p, clf_b1, bufO, N, H2, H4, 1, 0, stream);
    bn_kernel<<<(N * H4 + 255) / 256, 256, 0, stream>>>(bufO, clf_bn_g, clf_bn_b, N * H4, H4);
    clf_softmax_kernel<<<(N + 255) / 256, 256, 0, stream>>>(bufO, clf_W2, clf_b2, (float*)d_out, N);

    (void)in_sizes; (void)n_in; (void)out_size; (void)ws_size;
}
